// XdiffPerciever_13829794693599
// MI455X (gfx1250) — compile-verified, matrix operand order checked by reading, not run on hardware
//
#include <hip/hip_runtime.h>

// ---------------------------------------------------------------------------
// MI455X (gfx1250) implementation of the XdiffPerceiver forward pass.
// All dense math routed through V_WMMA_F32_16X16X32_BF16 (wave32 WMMA),
// fp32 accumulation. GEMM A-tiles staged with GLOBAL_LOAD_ASYNC_TO_LDS_B128
// (ASYNCcnt) and double-buffered so the DMA overlaps WMMA issue.
// Distance-feature attention bias computed on VALU/TRANS and fused into the
// logits kernel epilogue.
// ---------------------------------------------------------------------------

typedef __attribute__((ext_vector_type(16))) __bf16 v16bf;
typedef __attribute__((ext_vector_type(8)))  float  v8f;

union FragU { unsigned u[8]; v16bf v; };

__device__ __forceinline__ unsigned short f2bf(float f) {
    unsigned u = __builtin_bit_cast(unsigned, f);
    unsigned r = (u + 0x7FFFu + ((u >> 16) & 1u)) >> 16;   // round-to-nearest-even
    return (unsigned short)r;
}

__device__ __forceinline__ float gelu_f(float x) {
    return 0.5f * x * (1.0f + erff(x * 0.70710678118654752f));
}

// LDS byte offset of a __shared__ object: per CDNA5 aperture rules the low 32
// bits of a generic pointer into the LDS aperture ARE the LDS byte address.
__device__ __forceinline__ unsigned lds_off(const void* p) {
    return (unsigned)(unsigned long long)p;
}

// Async DMA: global (16B) -> LDS, tracked by ASYNCcnt (gfx1250).
__device__ __forceinline__ void async_copy_b128(unsigned lds_addr, const float* gptr) {
    asm volatile("global_load_async_to_lds_b128 %0, %1, off"
                 :: "v"(lds_addr), "v"(gptr) : "memory");
}
__device__ __forceinline__ void wait_async0() {
    asm volatile("s_wait_asynccnt 0" ::: "memory");
}

// Assemble a 16x32 bf16 WMMA A/B fragment from row-major bf16 LDS.
// Per CDNA5 ISA 7.12.2 (16-bit A 16x32): lanes 0-15 hold K{0..7,16..23},
// lanes 16-31 hold K{8..15,24..31}, two K values per VGPR.
__device__ __forceinline__ v16bf lds_frag(const unsigned short* p0, int rowstride) {
    int lane = threadIdx.x & 31;
    int half = lane >> 4;
    const unsigned short* p = p0 + (lane & 15) * rowstride;
    FragU f;
#pragma unroll
    for (int j = 0; j < 4; ++j) {
        f.u[j]     = *(const unsigned*)(p + half * 8 + 2 * j);
        f.u[4 + j] = *(const unsigned*)(p + 16 + half * 8 + 2 * j);
    }
    return f.v;
}

// Same fragment, but sourced from a raw fp32 LDS tile (async-copied);
// converts fp32 -> bf16 and packs while assembling.
__device__ __forceinline__ v16bf lds_frag_f32(const float* p0, int rowstride) {
    int lane = threadIdx.x & 31;
    int half = lane >> 4;
    const float* p = p0 + (lane & 15) * rowstride;
    FragU f;
#pragma unroll
    for (int j = 0; j < 4; ++j) {
        float2 lo = *(const float2*)(p + half * 8 + 2 * j);
        float2 hi = *(const float2*)(p + 16 + half * 8 + 2 * j);
        f.u[j]     = (unsigned)f2bf(lo.x) | ((unsigned)f2bf(lo.y) << 16);
        f.u[4 + j] = (unsigned)f2bf(hi.x) | ((unsigned)f2bf(hi.y) << 16);
    }
    return f.v;
}

#define WMMA_BF16(a, b, c) \
    __builtin_amdgcn_wmma_f32_16x16x32_bf16(false, (a), false, (b), (short)0, (c), false, false)

// ---------------------------------------------------------------------------
// Generic batched GEMM:  C = act(A @ W + bias) [+ residual]
// A (M,K) row-major (lda), W (K,N) row-major (ldb), C (M,N) (ldc), fp32 I/O.
// Batch z: b = z/Hn, h = z%Hn; per-operand offsets b*s?o + h*s?h.
// Tile: 64x64, K-step 32, 128 threads = 4 waves, each wave 32x32 (2x2 WMMA).
// A tile: async DMA to LDS (fp32, double-buffered).  W tile: load+cvt+ds_store.
// Requires M % 64 == 0 (true for every call in this network).
// ---------------------------------------------------------------------------
__global__ void __launch_bounds__(128)
gemm_bf16_wmma(const float* __restrict__ A, const float* __restrict__ W,
               const float* __restrict__ bias, const float* __restrict__ res,
               float* __restrict__ C, int M, int N, int K,
               int lda, int ldb, int ldc, int Hn,
               long long sAo, long long sAh, long long sBo, long long sBh,
               long long sCo, long long sCh, int act)
{
    __shared__ __align__(16) float          Af[2][64][32];   // async-staged A (fp32)
    __shared__ __align__(16) unsigned short Bs[64][32];      // [n][k] bf16 (W transposed)

    int z  = blockIdx.z;
    int bo = z / Hn, hi = z % Hn;
    A += bo * sAo + hi * sAh;
    W += bo * sBo + hi * sBh;
    long long coff = bo * sCo + hi * sCh;
    C += coff;
    if (res) res += coff;

    int tid = threadIdx.x;
    int m0 = blockIdx.y * 64, n0 = blockIdx.x * 64;

    v8f acc[2][2];
#pragma unroll
    for (int i = 0; i < 2; ++i)
#pragma unroll
        for (int j = 0; j < 2; ++j)
#pragma unroll
            for (int r = 0; r < 8; ++r) acc[i][j][r] = 0.0f;

    const unsigned afbase = lds_off(&Af[0][0][0]);
    // Each thread DMAs 4x16B: contiguous LDS, rows (tid>>3)+16p, cols (tid&7)*4.
    auto issueA = [&](int buf, int k0) {
        unsigned l0 = afbase + (unsigned)buf * 8192u + (unsigned)tid * 16u;
        const float* g = A + (long long)(m0 + (tid >> 3)) * lda + k0 + (tid & 7) * 4;
#pragma unroll
        for (int p2 = 0; p2 < 4; ++p2)
            async_copy_b128(l0 + (unsigned)p2 * 2048u, g + (long long)16 * p2 * lda);
    };

    issueA(0, 0);

    for (int k0 = 0; k0 < K; k0 += 32) {
        int buf = (k0 >> 5) & 1;
        {   // W tile 32x64 -> Bs[n][k] (transpose + fp32->bf16)
            int c = (tid & 15) * 4, r = tid >> 4;
#pragma unroll
            for (int p2 = 0; p2 < 4; ++p2) {
                int kk = r + 8 * p2;
                const float* wp = W + (long long)(k0 + kk) * ldb + n0 + c;
                float v0 = 0.f, v1 = 0.f, v2 = 0.f, v3 = 0.f;
                if (n0 + c + 3 < N) {
                    float4 v = *(const float4*)wp;
                    v0 = v.x; v1 = v.y; v2 = v.z; v3 = v.w;
                } else {
                    if (n0 + c + 0 < N) v0 = wp[0];
                    if (n0 + c + 1 < N) v1 = wp[1];
                    if (n0 + c + 2 < N) v2 = wp[2];
                    if (n0 + c + 3 < N) v3 = wp[3];
                }
                Bs[c + 0][kk] = f2bf(v0); Bs[c + 1][kk] = f2bf(v1);
                Bs[c + 2][kk] = f2bf(v2); Bs[c + 3][kk] = f2bf(v3);
            }
        }
        if (k0 + 32 < K)   // gfx1250 global_prefetch_b8 of next W tile
            __builtin_prefetch(W + (long long)(k0 + 32 + (tid >> 4)) * ldb + n0, 0, 1);

        wait_async0();       // this wave's A-tile DMA complete
        __syncthreads();     // all waves' DMA + Bs stores visible

        if (k0 + 32 < K) issueA(buf ^ 1, k0 + 32);   // overlap next DMA with WMMA

        int wave = tid >> 5;
        int wm = (wave >> 1) * 32, wn = (wave & 1) * 32;
        v16bf a0 = lds_frag_f32(&Af[buf][wm][0],      32);
        v16bf a1 = lds_frag_f32(&Af[buf][wm + 16][0], 32);
        v16bf b0 = lds_frag(&Bs[wn][0],      32);
        v16bf b1 = lds_frag(&Bs[wn + 16][0], 32);
        acc[0][0] = WMMA_BF16(a0, b0, acc[0][0]);
        acc[0][1] = WMMA_BF16(a0, b1, acc[0][1]);
        acc[1][0] = WMMA_BF16(a1, b0, acc[1][0]);
        acc[1][1] = WMMA_BF16(a1, b1, acc[1][1]);
        __syncthreads();     // protect Bs (and retired Af buffer) for next iter
    }

    // Epilogue.  C/D layout: VGPR r, lanes 0-15 -> (M=r, N=lane); 16-31 -> (M=8+r).
    int wave = tid >> 5, lane = tid & 31, half = lane >> 4, col = lane & 15;
    int wm = (wave >> 1) * 32, wn = (wave & 1) * 32;
#pragma unroll
    for (int i = 0; i < 2; ++i)
#pragma unroll
        for (int j = 0; j < 2; ++j)
#pragma unroll
            for (int r = 0; r < 8; ++r) {
                int m = m0 + wm + i * 16 + half * 8 + r;
                int n = n0 + wn + j * 16 + col;
                if (m < M && n < N) {
                    float v = acc[i][j][r];
                    if (bias) v += bias[n];
                    if (act == 1) v = gelu_f(v);
                    if (res) v += res[(long long)m * ldc + n];
                    C[(long long)m * ldc + n] = v;
                }
            }
}

// ---------------------------------------------------------------------------
// Attention logits:  L[z,q,k] = (Q_h . K_h)/8 + xdiff_bias/sqrt(26)
// Q,K are (B,S,512) slices (head h = cols h*64..h*64+63); xdiff features
// recomputed on the fly from positions (12 sincos per (q,k) pair).
// Grid: (Sk/64, Sq/32, B*H).  Block 128 = 4 waves, each 16q x 32k.
// ---------------------------------------------------------------------------
__global__ void __launch_bounds__(128)
attn_logits(const float* __restrict__ Q, const float* __restrict__ Kb,
            const float* __restrict__ XF, const float* __restrict__ Pq,
            const float* __restrict__ Pk, float* __restrict__ Lg,
            int Sq, int Sk, int kshift, int kN)
{
    __shared__ unsigned short Qs[32][64];
    __shared__ unsigned short Ks[64][64];
    __shared__ float Xs[32][26];

    int z = blockIdx.z, b = z >> 3, h = z & 7;
    int q0 = blockIdx.y * 32, k0 = blockIdx.x * 64;
    int tid = threadIdx.x;

    const float* Qp = Q  + ((long long)b * Sq + q0) * 512 + h * 64;
    const float* Kp = Kb + ((long long)b * Sk + k0) * 512 + h * 64;

    {   // load + convert Q (32x64) and K (64x64) tiles
        int c = (tid & 15) * 4, r = tid >> 4;
#pragma unroll
        for (int p2 = 0; p2 < 4; ++p2) {
            int row = r + 8 * p2;
            float4 v = *(const float4*)(Qp + (long long)row * 512 + c);
            Qs[row][c + 0] = f2bf(v.x); Qs[row][c + 1] = f2bf(v.y);
            Qs[row][c + 2] = f2bf(v.z); Qs[row][c + 3] = f2bf(v.w);
        }
#pragma unroll
        for (int p2 = 0; p2 < 8; ++p2) {
            int row = r + 8 * p2;
            float4 v = *(const float4*)(Kp + (long long)row * 512 + c);
            Ks[row][c + 0] = f2bf(v.x); Ks[row][c + 1] = f2bf(v.y);
            Ks[row][c + 2] = f2bf(v.z); Ks[row][c + 3] = f2bf(v.w);
        }
    }
    for (int i = tid; i < 32 * 26; i += 128) {
        int qr = i / 26, f = i % 26;
        Xs[qr][f] = XF[((long long)b * Sq + q0 + qr) * 208 + h * 26 + f];
    }
    __syncthreads();

    int wave = tid >> 5, lane = tid & 31, half = lane >> 4, col = lane & 15;
    int wm = (wave >> 1) * 16, wn = (wave & 1) * 32;

    v8f acc[2];
#pragma unroll
    for (int j = 0; j < 2; ++j)
#pragma unroll
        for (int r = 0; r < 8; ++r) acc[j][r] = 0.0f;

#pragma unroll
    for (int ks = 0; ks < 64; ks += 32) {   // depth 64 = 2 WMMA K-steps
        v16bf a  = lds_frag(&Qs[wm][0] + ks,      64);
        v16bf b0 = lds_frag(&Ks[wn][0] + ks,      64);
        v16bf b1 = lds_frag(&Ks[wn + 16][0] + ks, 64);
        acc[0] = WMMA_BF16(a, b0, acc[0]);
        acc[1] = WMMA_BF16(a, b1, acc[1]);
    }

    const float RSQD = 0.125f;            // 1/sqrt(64)
    const float RSQF = 0.19611613513818f; // 1/sqrt(26)
    const float PI   = 3.14159265358979f;
#pragma unroll
    for (int j = 0; j < 2; ++j)
#pragma unroll
        for (int r = 0; r < 8; ++r) {
            int ql = wm + half * 8 + r;          // local q row (0..31)
            int qg = q0 + ql;
            int kg = k0 + wn + j * 16 + col;
            const float* qp = Pq + ((long long)b * 64 + (qg >> 2)) * 3;
            const float* kp = Pk + ((long long)b * kN + (kg >> kshift)) * 3;
            float dx = qp[0] - kp[0], dy = qp[1] - kp[1], dz = qp[2] - kp[2];
            float d = sqrtf(dx * dx + dy * dy + dz * dz);
            float bias = Xs[ql][0] * d + Xs[ql][1] * d * d;
#pragma unroll
            for (int kk = 1; kk <= 12; ++kk) {
                float s, c;
                __sincosf(PI * (float)kk * d, &s, &c);
                bias += Xs[ql][2 * kk] * s + Xs[ql][2 * kk + 1] * c;
            }
            Lg[((long long)z * Sq + qg) * Sk + kg] = acc[j][r] * RSQD + bias * RSQF;
        }
}

// ---------------------------------------------------------------------------
// Row softmax over Sk (in place).  One block (128 threads, 4 wave32) per row.
// ---------------------------------------------------------------------------
__global__ void __launch_bounds__(128)
softmax_rows(float* __restrict__ Lg, int Sk)
{
    long long row = blockIdx.x;
    float* p = Lg + row * (long long)Sk;
    int tid = threadIdx.x;
    __shared__ float red[4];

    float mx = -3.4e38f;
    for (int i = tid; i < Sk; i += 128) mx = fmaxf(mx, p[i]);
#pragma unroll
    for (int o = 16; o > 0; o >>= 1) mx = fmaxf(mx, __shfl_xor(mx, o, 32));
    if ((tid & 31) == 0) red[tid >> 5] = mx;
    __syncthreads();
    mx = fmaxf(fmaxf(red[0], red[1]), fmaxf(red[2], red[3]));
    __syncthreads();

    float sum = 0.f;
    for (int i = tid; i < Sk; i += 128) { float e = __expf(p[i] - mx); p[i] = e; sum += e; }
#pragma unroll
    for (int o = 16; o > 0; o >>= 1) sum += __shfl_xor(sum, o, 32);
    if ((tid & 31) == 0) red[tid >> 5] = sum;
    __syncthreads();
    sum = red[0] + red[1] + red[2] + red[3];
    float inv = 1.0f / sum;
    for (int i = tid; i < Sk; i += 128) p[i] *= inv;
}

// ---------------------------------------------------------------------------
// LayerNorm over last dim D.  One block per row.
// ---------------------------------------------------------------------------
__global__ void __launch_bounds__(128)
layernorm_k(const float* __restrict__ X, const float* __restrict__ g,
            const float* __restrict__ bb, float* __restrict__ Y, int D)
{
    long long row = blockIdx.x;
    const float* x = X + row * (long long)D;
    float* y = Y + row * (long long)D;
    int tid = threadIdx.x;
    __shared__ float rs[4], rq[4];

    float s = 0.f, q = 0.f;
    for (int i = tid; i < D; i += 128) { float v = x[i]; s += v; q += v * v; }
#pragma unroll
    for (int o = 16; o > 0; o >>= 1) { s += __shfl_xor(s, o, 32); q += __shfl_xor(q, o, 32); }
    if ((tid & 31) == 0) { rs[tid >> 5] = s; rq[tid >> 5] = q; }
    __syncthreads();
    s = rs[0] + rs[1] + rs[2] + rs[3];
    q = rq[0] + rq[1] + rq[2] + rq[3];
    float mean = s / (float)D;
    float var  = q / (float)D - mean * mean;
    float inv  = rsqrtf(var + 1e-6f);
    for (int i = tid; i < D; i += 128) y[i] = (x[i] - mean) * inv * g[i] + bb[i];
}

// lat[b][q][c] = x_token[q % 4][c]
__global__ void init_lat_k(const float* __restrict__ tok, float* __restrict__ lat, int total)
{
    int i = blockIdx.x * 256 + threadIdx.x;
    if (i < total) {
        int c = i & 511;
        int q = (i >> 9) & 255;
        lat[i] = tok[((q & 3) << 9) + c];
    }
}

// ---------------------------------------------------------------------------
// Host-side orchestration
// ---------------------------------------------------------------------------
static void gemm(hipStream_t st, const float* A, const float* W, const float* bias,
                 const float* res, float* C, int M, int N, int K,
                 int lda, int ldb, int ldc, int act,
                 int batches = 1, int Hn = 1,
                 long long sAo = 0, long long sAh = 0,
                 long long sBo = 0, long long sBh = 0,
                 long long sCo = 0, long long sCh = 0)
{
    dim3 g((N + 63) / 64, (M + 63) / 64, batches);
    gemm_bf16_wmma<<<g, 128, 0, st>>>(A, W, bias, res, C, M, N, K, lda, ldb, ldc,
                                      Hn, sAo, sAh, sBo, sBh, sCo, sCh, act);
}

struct MhaW { const float *wq_w, *wq_b, *wk_w, *wk_b, *wv_w, *wv_b, *wx_w, *wx_b, *wo_w, *wo_b; };

struct Bufs {
    float *lat, *lnb, *qb, *attno, *xf, *big0, *big1, *big2, *lg;
};

static void run_mha(hipStream_t st, const MhaW& w, const float* qsrc, const float* kvsrc,
                    int Skv, const float* x_latents, const float* pk, int kshift, int kN,
                    float* kb, float* vb, const Bufs& B)
{
    gemm(st, qsrc,  w.wq_w, w.wq_b, nullptr, B.qb, 1024,    512, 512, 512, 512, 512, 0);
    gemm(st, kvsrc, w.wk_w, w.wk_b, nullptr, kb,   4 * Skv, 512, 512, 512, 512, 512, 0);
    gemm(st, kvsrc, w.wv_w, w.wv_b, nullptr, vb,   4 * Skv, 512, 512, 512, 512, 512, 0);
    gemm(st, B.qb,  w.wx_w, w.wx_b, nullptr, B.xf, 1024,    208, 512, 512, 208, 208, 0);

    dim3 gl(Skv / 64, 256 / 32, 4 * 8);
    attn_logits<<<gl, 128, 0, st>>>(B.qb, kb, B.xf, x_latents, pk, B.lg, 256, Skv, kshift, kN);
    softmax_rows<<<dim3(4 * 8 * 256), 128, 0, st>>>(B.lg, Skv);

    // out[b,:,h*64:...] = P(b,h) @ V_h(b):  batched over B*H with head strides
    gemm(st, B.lg, vb, nullptr, nullptr, B.attno, 256, 64, Skv, Skv, 512, 512, 0,
         32, 8,
         (long long)8 * 256 * Skv, (long long)256 * Skv,
         (long long)Skv * 512,     64,
         (long long)256 * 512,     64);

    gemm(st, B.attno, w.wo_w, w.wo_b, B.lat, B.lat, 1024, 512, 512, 512, 512, 512, 0);
}

static void run_enc(hipStream_t st, const float* const* e, const float* x_latents, const Bufs& B)
{
    layernorm_k<<<1024, 128, 0, st>>>(B.lat, e[5], e[4], B.lnb, 512);   // ln1
    MhaW w{e[13], e[12], e[9], e[8], e[15], e[14], e[17], e[16], e[11], e[10]};
    run_mha(st, w, B.lnb, B.lnb, 256, x_latents, x_latents, 2, 64, B.big2, B.big0, B);
    layernorm_k<<<1024, 128, 0, st>>>(B.lat, e[7], e[6], B.lnb, 512);   // ln2
    gemm(st, B.lnb, e[1], e[0], nullptr, B.big2, 1024, 2048, 512, 512, 2048, 2048, 1);
    gemm(st, B.big2, e[3], e[2], B.lat, B.lat, 1024, 512, 2048, 2048, 512, 512, 0);
}

static void run_cross(hipStream_t st, const float* const* c, const float* inputs,
                      const float* x_inputs, const float* x_latents, const Bufs& B)
{
    // inp = gelu(inputs @ inp_w + inp_b); inp = ln1b(inp)
    gemm(st, inputs, c[5], c[4], nullptr, B.big0, 4096, 512, 32, 32, 512, 512, 1);
    layernorm_k<<<4096, 128, 0, st>>>(B.big0, c[9], c[8], B.big1, 512); // ln1b
    layernorm_k<<<1024, 128, 0, st>>>(B.lat,  c[7], c[6], B.lnb, 512);  // ln1a
    MhaW w{c[17], c[16], c[13], c[12], c[19], c[18], c[21], c[20], c[15], c[14]};
    run_mha(st, w, B.lnb, B.big1, 1024, x_latents, x_inputs, 0, 1024, B.big2, B.big0, B);
    layernorm_k<<<1024, 128, 0, st>>>(B.lat, c[11], c[10], B.lnb, 512); // ln2
    gemm(st, B.lnb, c[1], c[0], nullptr, B.big2, 1024, 2048, 512, 512, 2048, 2048, 1);
    gemm(st, B.big2, c[3], c[2], B.lat, B.lat, 1024, 512, 2048, 2048, 512, 512, 0);
}

// params pytree leaf sizes in jax flatten order (dict keys sorted; Dn->{b,w}, LN->{b,g})
static int build_sizes(int* sz)
{
    int n = 0;
    auto dn  = [&](int di, int dout) { sz[n++] = dout; sz[n++] = di * dout; };
    auto ln  = [&](int d)            { sz[n++] = d;    sz[n++] = d; };
    auto mha = [&]() { dn(512, 512); dn(512, 512); dn(512, 512); dn(512, 512); dn(512, 208); }; // wk,wo,wq,wv,wx
    for (int r = 0; r < 2; ++r) {               // cross[0..1]
        dn(512, 2048); dn(2048, 512); dn(32, 512);
        ln(512); ln(512); ln(512); mha();
    }
    for (int g = 0; g < 3; ++g)                 // enc[0..2][0..1]
        for (int l = 0; l < 2; ++l) {
            dn(512, 2048); dn(2048, 512); ln(512); ln(512); mha();
        }
    dn(512, 1);        // final
    ln(512);           // ln_f
    dn(512, 512);      // pre_final[0]
    sz[n++] = 4 * 512; // x_token
    return n;          // 159
}

extern "C" void kernel_launch(void* const* d_in, const int* in_sizes, int n_in,
                              void* d_out, int out_size, void* d_ws, size_t ws_size,
                              hipStream_t stream)
{
    (void)in_sizes; (void)out_size; (void)ws_size;
    const float* inputs    = (const float*)d_in[0];   // (4,1024,32)
    const float* x_inputs  = (const float*)d_in[1];   // (4,1024,3)
    const float* x_latents = (const float*)d_in[2];   // (4,64,3)

    int sizes[160];
    int nl = build_sizes(sizes);
    const float* L[160];
    if (n_in >= 3 + nl) {
        for (int i = 0; i < nl; ++i) L[i] = (const float*)d_in[3 + i];
    } else {
        const float* p = (const float*)d_in[3];       // packed leaves
        for (int i = 0; i < nl; ++i) { L[i] = p; p += sizes[i]; }
    }

    float* ws = (float*)d_ws;
    Bufs B;
    B.lat   = ws;                    // 4*256*512
    B.lnb   = B.lat   + 524288;
    B.qb    = B.lnb   + 524288;
    B.attno = B.qb    + 524288;
    B.xf    = B.attno + 524288;      // 4*256*208 (padded to 262144)
    B.big0  = B.xf    + 262144;      // 4*1024*512 : inp_act / V
    B.big1  = B.big0  + 2097152;     // inp_norm
    B.big2  = B.big1  + 2097152;     // K / FFN hidden
    B.lg    = B.big2  + 2097152;     // 4*8*256*1024 logits

    init_lat_k<<<2048, 256, 0, stream>>>(L[158], B.lat, 524288);

    for (int r = 0; r < 2; ++r) {
        run_cross(stream, L + r * 22, inputs, x_inputs, x_latents, B);
        for (int l = 0; l < 2; ++l)
            run_enc(stream, L + 44 + (r * 2 + l) * 18, x_latents, B);
    }
    for (int l = 0; l < 2; ++l)
        run_enc(stream, L + 44 + (4 + l) * 18, x_latents, B);

    layernorm_k<<<1024, 128, 0, stream>>>(B.lat, L[155], L[154], B.lnb, 512); // ln_f
    gemm(stream, B.lnb, L[157], L[156], nullptr, B.qb, 1024, 512, 512, 512, 512, 512, 1); // pre_final + gelu
    gemm(stream, B.qb, L[153], L[152], nullptr, (float*)d_out, 1024, 1, 512, 512, 1, 1, 0); // final
}
